// Attention_73675868996404
// MI455X (gfx1250) — matrix operand order repaired and checked by reading, not compiled, not run on hardware
//
#include <hip/hip_runtime.h>

typedef __attribute__((ext_vector_type(16))) __bf16 v16bf;
typedef __attribute__((ext_vector_type(8)))  float  v8f;
typedef __attribute__((ext_vector_type(4)))  unsigned int u32x4;   // plain 128-bit vec

#define AS3 __attribute__((address_space(3)))
#define AS1 __attribute__((address_space(1)))

union Frag16 {
    v16bf v;
    u32x4 q[2];
};

#define LSEQ 2048
#define HD   64
#define DIMS 1024
#define NQKV 3072
#define MTOT 8192        // B*L
#define SK   40          // LDS row stride (halfwords): 80B, 16B-aligned, bank-spread
#define AS_ELEMS (128 * SK)   // one A buffer (128x32 tile)
#define BS_ELEMS (256 * SK)   // one B buffer (256x32 tile, stored [n][k])

__device__ __forceinline__ unsigned short f32_to_bf16(float f) {
    unsigned int u = __float_as_uint(f);
    unsigned int r = u + 0x7FFFu + ((u >> 16) & 1u);
    return (unsigned short)(r >> 16);
}

__device__ __forceinline__ float swish_f(float v) {
    return v / (1.0f + __expf(-v));
}

// ---- address-space helpers ------------------------------------------------
__device__ __forceinline__ const AS3 unsigned short* to_lds(const unsigned short* p) {
    return (const AS3 unsigned short*)p;
}
__device__ __forceinline__ AS3 unsigned short* to_lds_mut(unsigned short* p) {
    return (AS3 unsigned short*)p;
}
__device__ __forceinline__ unsigned int lds_off(const unsigned short* p) {
    return (unsigned int)(unsigned long long)to_lds(p);
}
__device__ __forceinline__ u32x4 ld_lds128(const AS3 unsigned short* p) {
    return *(const AS3 u32x4*)p;       // ds_load_b128
}
__device__ __forceinline__ u32x4 ld_g128(const unsigned short* p) {
    return *(const AS1 u32x4*)(const AS1 unsigned short*)p;   // global_load_b128
}

// CDNA5 async copy: global -> LDS, 16B per lane, tracked by ASYNCcnt
__device__ __forceinline__ void async_b128(unsigned int lds_byte_off, const void* gaddr) {
    asm volatile("global_load_async_to_lds_b128 %0, %1, off"
                 :: "v"(lds_byte_off), "v"(gaddr) : "memory");
}
__device__ __forceinline__ void wait_asynccnt0() {
    asm volatile("s_wait_asynccnt 0" ::: "memory");
}

// ---------------------------------------------------------------------------
// f32 -> bf16 conversion (grid-stride)
// ---------------------------------------------------------------------------
__global__ void cvt_bf16_kernel(const float* __restrict__ src,
                                unsigned short* __restrict__ dst, int n) {
    int i = blockIdx.x * blockDim.x + threadIdx.x;
    int stride = gridDim.x * blockDim.x;
    for (; i < n; i += stride) dst[i] = f32_to_bf16(src[i]);
}

// ---------------------------------------------------------------------------
// Weight transpose+convert: src f32 [K][N] -> dst bf16 [N][K] (32x32 LDS tiles)
// ---------------------------------------------------------------------------
__global__ __launch_bounds__(256) void cvt_transpose_kernel(
    const float* __restrict__ src, unsigned short* __restrict__ dst, int K, int N) {
    __shared__ float tile[32][33];
    const int nb = blockIdx.x * 32, kb = blockIdx.y * 32;
    const int tx = threadIdx.x & 31, ty = threadIdx.x >> 5;   // ty 0..7
#pragma unroll
    for (int i = 0; i < 32; i += 8)
        tile[ty + i][tx] = src[(size_t)(kb + ty + i) * N + nb + tx];
    __syncthreads();
#pragma unroll
    for (int i = 0; i < 32; i += 8)
        dst[(size_t)(nb + ty + i) * K + kb + tx] = f32_to_bf16(tile[tx][ty + i]);
}

// ---------------------------------------------------------------------------
// Async tile issue: A tile 128x32 ([m][k]) and B tile 256x32 ([n][k], from
// pre-transposed weights). All copies are contiguous-K b128 async loads.
// ---------------------------------------------------------------------------
__device__ __forceinline__ void issue_tile(const unsigned short* __restrict__ A,
                                           const unsigned short* __restrict__ Bt,
                                           int lda, int ldbt, int row0, int col0,
                                           int kb, unsigned int as_base,
                                           unsigned int bs_base, int tid) {
    {   // A: 2 x b128 per thread
        const int mm = tid >> 1;
        const int k8 = (tid & 1) * 8;
#pragma unroll
        for (int rep = 0; rep < 2; ++rep) {
            const int kk = k8 + rep * 16;
            async_b128(as_base + (unsigned)(mm * SK + kk) * 2,
                       A + (size_t)(row0 + mm) * lda + kb + kk);
        }
    }
    {   // B: 4 x b128 per thread (row n = tid)
        const int n = tid;
#pragma unroll
        for (int i = 0; i < 4; ++i) {
            const int kk = i * 8;
            async_b128(bs_base + (unsigned)(n * SK + kk) * 2,
                       Bt + (size_t)(col0 + n) * ldbt + kb + kk);
        }
    }
}

// ---------------------------------------------------------------------------
// GEMM mainloop: C(128x256) += A(128xK) * Bt(NxK)^T, bf16 in / f32 acc.
// 8 waves: 2 along M x 4 along N; wave tile 64x64 = 4x4 WMMA fragments.
// Double-buffered LDS fed by async-to-LDS copies; 1 barrier per k-step.
// All fragment reads through explicit AS(3) pointers -> ds_load_b128.
// ---------------------------------------------------------------------------
__device__ __forceinline__ void gemm_mainloop(const unsigned short* __restrict__ A,
                                              const unsigned short* __restrict__ Bt,
                                              int K, int row0, int col0,
                                              unsigned short* SMEM, v8f acc[4][4]) {
    const int tid  = threadIdx.x;
    const int lane = tid & 31;
    const int wid  = tid >> 5;
    const int wm   = wid & 1;
    const int wn   = wid >> 1;
    const int ln16 = lane & 15;
    const int lhi  = lane >> 4;

    const AS3 unsigned short* lds = to_lds(SMEM);
    const unsigned int smem_base = lds_off(SMEM);
    const unsigned int as_base[2] = {smem_base, smem_base + AS_ELEMS * 2u};
    const unsigned int bs_base[2] = {smem_base + 4u * AS_ELEMS,
                                     smem_base + 4u * AS_ELEMS + BS_ELEMS * 2u};

    issue_tile(A, Bt, K, K, row0, col0, 0, as_base[0], bs_base[0], tid);
    wait_asynccnt0();
    __syncthreads();

    const int T = K / 32;
    for (int t = 0; t < T; ++t) {
        const int cur = t & 1;
        if (t + 1 < T)
            issue_tile(A, Bt, K, K, row0, col0, (t + 1) * 32,
                       as_base[cur ^ 1], bs_base[cur ^ 1], tid);

        // buffer select via index math on a single AS(3) base (keeps ds path)
        const AS3 unsigned short* Asc = lds + cur * AS_ELEMS;
        const AS3 unsigned short* Bsc = lds + 2 * AS_ELEMS + cur * BS_ELEMS;

        Frag16 af[4];
#pragma unroll
        for (int fm = 0; fm < 4; ++fm) {
            // A frag: lanes 0-15 K 0-15; lanes 16-31 K 16-31
            const AS3 unsigned short* base =
                Asc + (wm * 64 + fm * 16 + ln16) * SK + lhi * 16;
            af[fm].q[0] = ld_lds128(base);
            af[fm].q[1] = ld_lds128(base + 8);
        }
#pragma unroll
        for (int fn = 0; fn < 4; ++fn) {
            // B frag: lanes 0-15 K 0-15; lanes 16-31 K 16-31
            const AS3 unsigned short* base =
                Bsc + (wn * 64 + fn * 16 + ln16) * SK + lhi * 16;
            Frag16 bf;
            bf.q[0] = ld_lds128(base);
            bf.q[1] = ld_lds128(base + 8);
#pragma unroll
            for (int fm = 0; fm < 4; ++fm)
                acc[fm][fn] = __builtin_amdgcn_wmma_f32_16x16x32_bf16(
                    false, af[fm].v, false, bf.v, (short)0, acc[fm][fn],
                    false, false);
        }
        if (t + 1 < T) wait_asynccnt0();
        __syncthreads();
    }
}

// ---------------------------------------------------------------------------
// GEMM1: qkv = swish(x @ W_fc + b_fc); scatter q,k ([BH,L,64] bf16) and
// v transposed ([BH,64,L] bf16). W_fc is pre-transposed [N][K].
// ---------------------------------------------------------------------------
__global__ __launch_bounds__(256) void gemm_qkv_kernel(
    const unsigned short* __restrict__ xb, const unsigned short* __restrict__ wfcT,
    const float* __restrict__ bias,
    unsigned short* __restrict__ qb, unsigned short* __restrict__ kb,
    unsigned short* __restrict__ vT) {
    __shared__ unsigned short SMEM[2 * AS_ELEMS + 2 * BS_ELEMS];
    v8f acc[4][4] = {};

    const int row0 = blockIdx.y * 128;
    const int col0 = blockIdx.x * 256;
    gemm_mainloop(xb, wfcT, DIMS, row0, col0, SMEM, acc);

    const int lane = threadIdx.x & 31, wid = threadIdx.x >> 5;
    const int wm = wid & 1, wn = wid >> 1, ln16 = lane & 15, lhi = lane >> 4;
#pragma unroll
    for (int fm = 0; fm < 4; ++fm)
#pragma unroll
        for (int fn = 0; fn < 4; ++fn)
#pragma unroll
            for (int r = 0; r < 8; ++r) {
                int grow = row0 + wm * 64 + fm * 16 + r + 8 * lhi;
                int gcol = col0 + wn * 64 + fn * 16 + ln16;
                float v = acc[fm][fn][r] + bias[gcol];
                unsigned short bv = f32_to_bf16(swish_f(v));
                int bidx = grow >> 11;            // / L
                int l    = grow & (LSEQ - 1);
                int h    = gcol / 192;
                int rem  = gcol - h * 192;
                int part = rem >> 6;
                int d    = rem & 63;
                size_t bh = (size_t)(bidx * 16 + h);
                if (part == 0)      qb[(bh * LSEQ + l) * HD + d] = bv;
                else if (part == 1) kb[(bh * LSEQ + l) * HD + d] = bv;
                else                vT[(bh * HD + d) * LSEQ + l] = bv;
            }
}

// ---------------------------------------------------------------------------
// Flash attention: one wave = 16 query rows, online softmax over 64-key blocks
// ---------------------------------------------------------------------------
#define PROW 72   // P staging row stride (halfwords): 144B, 16B aligned
__global__ __launch_bounds__(256) void attn_kernel(
    const unsigned short* __restrict__ qb, const unsigned short* __restrict__ kb,
    const unsigned short* __restrict__ vT, unsigned short* __restrict__ newv) {
    __shared__ unsigned short Pl[8][16 * PROW];

    const int lane = threadIdx.x & 31, wid = threadIdx.x >> 5;
    const int ln16 = lane & 15, lhi = lane >> 4;
    const int b = blockIdx.z, h = blockIdx.y;
    const int row0 = blockIdx.x * 128 + wid * 16;   // query row base in L

    const unsigned short* qh = qb + ((size_t)(b * 16 + h) * LSEQ) * HD;
    const unsigned short* kh = kb + ((size_t)(b * 16 + h) * LSEQ) * HD;
    const unsigned short* vh = vT + ((size_t)(b * 16 + h) * HD) * LSEQ;
    AS3 unsigned short* Pw = to_lds_mut(&Pl[wid][0]);

    // Q A-fragments (resident): hd=64 -> two K=32 chunks
    Frag16 qa[2];
#pragma unroll
    for (int kc = 0; kc < 2; ++kc) {
        const unsigned short* base = qh + (size_t)(row0 + ln16) * HD + kc * 32 + lhi * 16;
        qa[kc].q[0] = ld_g128(base);
        qa[kc].q[1] = ld_g128(base + 8);
    }

    float rmax[8], rsum[8];
#pragma unroll
    for (int r = 0; r < 8; ++r) { rmax[r] = -1e30f; rsum[r] = 0.0f; }
    v8f o[4] = {};

    for (int j = 0; j < LSEQ; j += 64) {
        // S = Q @ K^T  (16 x 64 keys)
        v8f s[4] = {};
#pragma unroll
        for (int nf = 0; nf < 4; ++nf) {
#pragma unroll
            for (int kc = 0; kc < 2; ++kc) {
                Frag16 kf;
                const unsigned short* base =
                    kh + (size_t)(j + nf * 16 + ln16) * HD + kc * 32 + lhi * 16;
                kf.q[0] = ld_g128(base);
                kf.q[1] = ld_g128(base + 8);
                s[nf] = __builtin_amdgcn_wmma_f32_16x16x32_bf16(
                    false, qa[kc].v, false, kf.v, (short)0, s[nf], false, false);
            }
        }
        // online softmax update per row (VGPR r <-> row r+8*lhi)
#pragma unroll
        for (int r = 0; r < 8; ++r) {
            float mv = fmaxf(fmaxf(s[0][r], s[1][r]), fmaxf(s[2][r], s[3][r]));
            mv = fmaxf(mv, __shfl_xor(mv, 1, 32));
            mv = fmaxf(mv, __shfl_xor(mv, 2, 32));
            mv = fmaxf(mv, __shfl_xor(mv, 4, 32));
            mv = fmaxf(mv, __shfl_xor(mv, 8, 32));
            float mnew = fmaxf(rmax[r], mv);
            float corr = __expf(rmax[r] - mnew);
            rmax[r] = mnew;
            rsum[r] *= corr;
#pragma unroll
            for (int nf = 0; nf < 4; ++nf) o[nf][r] *= corr;
            float psum = 0.0f;
#pragma unroll
            for (int nf = 0; nf < 4; ++nf) {
                float p = __expf(s[nf][r] - mnew);
                psum += p;
                Pw[(r + 8 * lhi) * PROW + nf * 16 + ln16] = f32_to_bf16(p);
            }
            psum += __shfl_xor(psum, 1, 32);
            psum += __shfl_xor(psum, 2, 32);
            psum += __shfl_xor(psum, 4, 32);
            psum += __shfl_xor(psum, 8, 32);
            rsum[r] += psum;
        }
        // O += P @ V   (DS ops are in-order within the wave; no barrier needed)
#pragma unroll
        for (int kc2 = 0; kc2 < 2; ++kc2) {
            Frag16 pa;
            const AS3 unsigned short* pb = Pw + ln16 * PROW + kc2 * 32 + lhi * 16;
            pa.q[0] = ld_lds128(pb);
            pa.q[1] = ld_lds128(pb + 8);
#pragma unroll
            for (int nf = 0; nf < 4; ++nf) {
                Frag16 vb;
                const unsigned short* base =
                    vh + (size_t)(nf * 16 + ln16) * LSEQ + j + kc2 * 32 + lhi * 16;
                vb.q[0] = ld_g128(base);
                vb.q[1] = ld_g128(base + 8);
                o[nf] = __builtin_amdgcn_wmma_f32_16x16x32_bf16(
                    false, pa.v, false, vb.v, (short)0, o[nf], false, false);
            }
        }
    }
    // finalize and write new_v as bf16 [B*L, H*64]
#pragma unroll
    for (int r = 0; r < 8; ++r) {
        float inv = 1.0f / rsum[r];
        int lr = row0 + r + 8 * lhi;
#pragma unroll
        for (int nf = 0; nf < 4; ++nf) {
            size_t idx = (size_t)(b * LSEQ + lr) * DIMS + h * HD + nf * 16 + ln16;
            newv[idx] = f32_to_bf16(o[nf][r] * inv);
        }
    }
}

// ---------------------------------------------------------------------------
// GEMM3: tmp = swish(new_v @ W_out + b_out) + x   (f32 out); W_out pre-transposed
// ---------------------------------------------------------------------------
__global__ __launch_bounds__(256) void gemm_out_kernel(
    const unsigned short* __restrict__ nv, const unsigned short* __restrict__ woutT,
    const float* __restrict__ bias, const float* __restrict__ xres,
    float* __restrict__ tmp) {
    __shared__ unsigned short SMEM[2 * AS_ELEMS + 2 * BS_ELEMS];
    v8f acc[4][4] = {};

    const int row0 = blockIdx.y * 128;
    const int col0 = blockIdx.x * 256;
    gemm_mainloop(nv, woutT, DIMS, row0, col0, SMEM, acc);

    const int lane = threadIdx.x & 31, wid = threadIdx.x >> 5;
    const int wm = wid & 1, wn = wid >> 1, ln16 = lane & 15, lhi = lane >> 4;
#pragma unroll
    for (int fm = 0; fm < 4; ++fm)
#pragma unroll
        for (int fn = 0; fn < 4; ++fn)
#pragma unroll
            for (int r = 0; r < 8; ++r) {
                int grow = row0 + wm * 64 + fm * 16 + r + 8 * lhi;
                int gcol = col0 + wn * 64 + fn * 16 + ln16;
                size_t idx = (size_t)grow * DIMS + gcol;
                tmp[idx] = swish_f(acc[fm][fn][r] + bias[gcol]) + xres[idx];
            }
}

// ---------------------------------------------------------------------------
// LayerNorm over last dim (1024), one block per row
// ---------------------------------------------------------------------------
__global__ __launch_bounds__(256) void ln_kernel(const float* __restrict__ tmp,
                                                 float* __restrict__ out) {
    __shared__ float rs[256], rq[256];
    const int row = blockIdx.x, tid = threadIdx.x;
    const float* p = tmp + (size_t)row * DIMS;
    float vals[4], s = 0.0f, ss = 0.0f;
#pragma unroll
    for (int i = 0; i < 4; ++i) {
        vals[i] = p[tid + i * 256];
        s += vals[i];
        ss += vals[i] * vals[i];
    }
    rs[tid] = s; rq[tid] = ss;
    __syncthreads();
    for (int off = 128; off > 0; off >>= 1) {
        if (tid < off) { rs[tid] += rs[tid + off]; rq[tid] += rq[tid + off]; }
        __syncthreads();
    }
    float mu = rs[0] * (1.0f / DIMS);
    float var = rq[0] * (1.0f / DIMS) - mu * mu;
    float inv = rsqrtf(var + 1e-5f);
    float* q = out + (size_t)row * DIMS;
#pragma unroll
    for (int i = 0; i < 4; ++i) q[tid + i * 256] = (vals[i] - mu) * inv;
}

// ---------------------------------------------------------------------------
// Host launcher
// ---------------------------------------------------------------------------
extern "C" void kernel_launch(void* const* d_in, const int* in_sizes, int n_in,
                              void* d_out, int out_size, void* d_ws, size_t ws_size,
                              hipStream_t stream) {
    (void)in_sizes; (void)n_in; (void)out_size; (void)ws_size;
    const float* x     = (const float*)d_in[0];
    const float* W_fc  = (const float*)d_in[1];
    const float* b_fc  = (const float*)d_in[2];
    const float* W_out = (const float*)d_in[3];
    const float* b_out = (const float*)d_in[4];
    float* out = (float*)d_out;

    char* ws = (char*)d_ws;
    // Workspace layout (72 MB, with aliasing):
    //  [0,16M)       xb (bf16 x)      -- reused as newv after GEMM1
    //  [16M,22M)   wfcT  (bf16, [N][K])
    //  [22M,24M) woutT (bf16, [N][K])
    //  [24M..)       qb(16M) kb(16M) vT(16M); tmpf(f32,32M) aliases qb+kb after attn
    unsigned short* xb    = (unsigned short*)(ws);
    unsigned short* wfcT  = (unsigned short*)(ws + 16777216);
    unsigned short* woutT = (unsigned short*)(ws + 16777216 + 6291456);
    unsigned short* qb    = (unsigned short*)(ws + 25165824);
    unsigned short* kb    = qb + 8388608;
    unsigned short* vT    = kb + 8388608;
    unsigned short* newv  = xb;         // alias: xb dead after GEMM1
    float* tmpf = (float*)qb;           // alias: q/k dead after attention

    cvt_bf16_kernel<<<4096, 256, 0, stream>>>(x, xb, MTOT * DIMS);
    cvt_transpose_kernel<<<dim3(NQKV / 32, DIMS / 32), 256, 0, stream>>>(
        W_fc, wfcT, DIMS, NQKV);
    cvt_transpose_kernel<<<dim3(DIMS / 32, DIMS / 32), 256, 0, stream>>>(
        W_out, woutT, DIMS, DIMS);

    gemm_qkv_kernel<<<dim3(NQKV / 256, MTOT / 128), 256, 0, stream>>>(
        xb, wfcT, b_fc, qb, kb, vT);

    attn_kernel<<<dim3(LSEQ / 128, 16, 4), 256, 0, stream>>>(qb, kb, vT, newv);

    gemm_out_kernel<<<dim3(DIMS / 256, MTOT / 128), 256, 0, stream>>>(
        newv, woutT, b_out, x, tmpf);

    ln_kernel<<<MTOT, 256, 0, stream>>>(tmpf, out);
}
